// SpeakerMemory_57028575756935
// MI455X (gfx1250) — matrix operand order repaired and checked by reading, not compiled, not run on hardware
//
#include <hip/hip_runtime.h>
#include <hip/hip_bf16.h>

// ---------------------------------------------------------------------------
// SpeakerMemory GRU update, MI455X (gfx1250, wave32, WMMA bf16)
// ---------------------------------------------------------------------------

typedef __attribute__((ext_vector_type(4)))  __bf16 v4bf;
typedef __attribute__((ext_vector_type(8)))  __bf16 v8bf;
typedef __attribute__((ext_vector_type(16))) __bf16 v16bf;
typedef __attribute__((ext_vector_type(4)))  float  v4f;
typedef __attribute__((ext_vector_type(8)))  float  v8f;

union BF16x16 { v16bf v; v8bf h[2]; };

__device__ __forceinline__ float sigmoid_f(float x) {
    return 1.0f / (1.0f + __expf(-x));
}

// ---------------------------------------------------------------------------
// last_pos[s] = max t with ids[t]==s, else -1
// ---------------------------------------------------------------------------
__global__ void init_lastpos_kernel(int* __restrict__ lp, int S) {
    int i = blockIdx.x * blockDim.x + threadIdx.x;
    if (i < S) lp[i] = -1;
}

__global__ void scatter_lastpos_kernel(const int* __restrict__ ids,
                                       int* __restrict__ lp, int T) {
    int t = blockIdx.x * blockDim.x + threadIdx.x;
    if (t < T) atomicMax(&lp[ids[t]], t);
}

// ---------------------------------------------------------------------------
// f32 -> bf16 conversion (flat, n multiple of 8)
// ---------------------------------------------------------------------------
__global__ void cvt_f32_to_bf16_kernel(const float* __restrict__ src,
                                       __bf16* __restrict__ dst, int n8) {
    int i = blockIdx.x * blockDim.x + threadIdx.x;
    if (i >= n8) return;
    const v4f* s = (const v4f*)src + (size_t)i * 2;
    v4f a = s[0], b = s[1];
    v8bf o;
    o[0] = (__bf16)a[0]; o[1] = (__bf16)a[1]; o[2] = (__bf16)a[2]; o[3] = (__bf16)a[3];
    o[4] = (__bf16)b[0]; o[5] = (__bf16)b[1]; o[6] = (__bf16)b[2]; o[7] = (__bf16)b[3];
    ((v8bf*)dst)[i] = o;
}

// ---------------------------------------------------------------------------
// Hbf[t,:] = bf16(speaker_memory[ids[t],:])   (gather + convert)
// ---------------------------------------------------------------------------
__global__ void gather_h_bf16_kernel(const float* __restrict__ mem,
                                     const int* __restrict__ ids,
                                     __bf16* __restrict__ dst, int H) {
    int t = blockIdx.x;
    size_t srcBase = (size_t)ids[t] * (size_t)H;
    size_t dstBase = (size_t)t * (size_t)H;
    for (int c = threadIdx.x * 4; c < H; c += blockDim.x * 4) {
        v4f a = *(const v4f*)(mem + srcBase + c);
        v4bf o;
        o[0] = (__bf16)a[0]; o[1] = (__bf16)a[1];
        o[2] = (__bf16)a[2]; o[3] = (__bf16)a[3];
        *(v4bf*)(dst + dstBase + c) = o;
    }
}

// ---------------------------------------------------------------------------
// Fused dual-GEMM + GRU gate + scatter-write kernel.
//   grid = (H/64, T/32), block = 192 (6 wave32's)
//   wave w in [0,6): region w of {i_r,i_z,i_n,h_r,h_z,h_n}
//   Each wave: 32x64 f32 tile = 2x4 WMMA_F32_16x16x32_BF16 tiles, K=H loop.
// ---------------------------------------------------------------------------
__global__ __launch_bounds__(192)
void gru_fused_kernel(const __bf16* __restrict__ Xbf,     // [T,H] bf16
                      const __bf16* __restrict__ Hbf,     // [T,H] bf16
                      const __bf16* __restrict__ Wihbf,   // [3H,H] bf16
                      const __bf16* __restrict__ Whhbf,   // [3H,H] bf16
                      const float*  __restrict__ b_ih,    // [3H]
                      const float*  __restrict__ b_hh,    // [3H]
                      const float*  __restrict__ speaker_memory, // [S,H] f32
                      const int*    __restrict__ ids,     // [T]
                      const int*    __restrict__ last_pos,// [S]
                      float*        __restrict__ out,     // [S,H] f32
                      int H)
{
    constexpr int M_TILE = 32;
    constexpr int N_TILE = 64;
    constexpr int LS     = 65;   // padded LDS row stride (floats)

    __shared__ float smem[6][M_TILE][LS];   // 49,920 B

    const int wave  = threadIdx.x >> 5;     // 0..5  (uniform per wave)
    const int lane  = threadIdx.x & 31;
    const int l16   = lane & 15;
    const int khalf = lane >> 4;            // 0 or 1

    const int colBase = blockIdx.x * N_TILE;    // within [0,H)
    const int rowBase = blockIdx.y * M_TILE;    // within [0,T)

    const __bf16* A = (wave < 3) ? Xbf   : Hbf;
    const __bf16* W = (wave < 3) ? Wihbf : Whhbf;
    const int wrowOff = (wave % 3) * H;         // gate region row offset in W

    v8f acc[2][4];
    const v8f vzero = {0.f, 0.f, 0.f, 0.f, 0.f, 0.f, 0.f, 0.f};
#pragma unroll
    for (int ms = 0; ms < 2; ++ms)
#pragma unroll
        for (int ns = 0; ns < 4; ++ns) acc[ms][ns] = vzero;

    // A fragment: contiguous K per lane-half (matches B):
    //   lane half 0 -> K = kb+0..15 ; half 1 -> kb+16..31
    const int koffA = khalf * 16;
    // B fragment: contiguous K per lane-half (same order as A):
    //   lane half 0 -> K = kb+0..15 ; half 1 -> kb+16..31
    const int koffB = khalf * 16;

    const size_t aRow0 = (size_t)(rowBase + l16) * (size_t)H;
    const size_t aRow1 = (size_t)(rowBase + 16 + l16) * (size_t)H;
    const size_t bRowBase = (size_t)(wrowOff + colBase + l16) * (size_t)H;

#pragma unroll 2
    for (int kb = 0; kb < H; kb += 32) {
        BF16x16 a0, a1;
        a0.h[0] = *(const v8bf*)(A + aRow0 + kb + koffA);
        a0.h[1] = *(const v8bf*)(A + aRow0 + kb + koffA + 8);
        a1.h[0] = *(const v8bf*)(A + aRow1 + kb + koffA);
        a1.h[1] = *(const v8bf*)(A + aRow1 + kb + koffA + 8);

        v16bf b0 = *(const v16bf*)(W + bRowBase + (size_t)0  * H + kb + koffB);
        v16bf b1 = *(const v16bf*)(W + bRowBase + (size_t)16 * H + kb + koffB);
        v16bf b2 = *(const v16bf*)(W + bRowBase + (size_t)32 * H + kb + koffB);
        v16bf b3 = *(const v16bf*)(W + bRowBase + (size_t)48 * H + kb + koffB);

        acc[0][0] = __builtin_amdgcn_wmma_f32_16x16x32_bf16(false, a0.v, false, b0, (short)0, acc[0][0], false, false);
        acc[0][1] = __builtin_amdgcn_wmma_f32_16x16x32_bf16(false, a0.v, false, b1, (short)0, acc[0][1], false, false);
        acc[0][2] = __builtin_amdgcn_wmma_f32_16x16x32_bf16(false, a0.v, false, b2, (short)0, acc[0][2], false, false);
        acc[0][3] = __builtin_amdgcn_wmma_f32_16x16x32_bf16(false, a0.v, false, b3, (short)0, acc[0][3], false, false);
        acc[1][0] = __builtin_amdgcn_wmma_f32_16x16x32_bf16(false, a1.v, false, b0, (short)0, acc[1][0], false, false);
        acc[1][1] = __builtin_amdgcn_wmma_f32_16x16x32_bf16(false, a1.v, false, b1, (short)0, acc[1][1], false, false);
        acc[1][2] = __builtin_amdgcn_wmma_f32_16x16x32_bf16(false, a1.v, false, b2, (short)0, acc[1][2], false, false);
        acc[1][3] = __builtin_amdgcn_wmma_f32_16x16x32_bf16(false, a1.v, false, b3, (short)0, acc[1][3], false, false);
    }

    // Spill accumulators to LDS per the 16x16 f32 C/D layout:
    //   VGPR v, lane-half k -> row v + 8*k ; column = lane%16
#pragma unroll
    for (int ms = 0; ms < 2; ++ms)
#pragma unroll
        for (int ns = 0; ns < 4; ++ns)
#pragma unroll
            for (int v = 0; v < 8; ++v) {
                int rl = ms * 16 + v + khalf * 8;
                int cl = ns * 16 + l16;
                smem[wave][rl][cl] = acc[ms][ns][v];
            }

    __syncthreads();

    // Elementwise GRU in f32, scatter only last occurrences.
    for (int idx = threadIdx.x; idx < M_TILE * N_TILE; idx += 192) {
        int rl = idx >> 6;          // local row (t offset)
        int cl = idx & 63;          // local col (hidden dim offset)
        int t  = rowBase + rl;
        int c  = colBase + cl;

        float i_r = smem[0][rl][cl] + b_ih[c];
        float i_z = smem[1][rl][cl] + b_ih[H + c];
        float i_n = smem[2][rl][cl] + b_ih[2 * H + c];
        float h_r = smem[3][rl][cl] + b_hh[c];
        float h_z = smem[4][rl][cl] + b_hh[H + c];
        float h_n = smem[5][rl][cl] + b_hh[2 * H + c];

        float r = sigmoid_f(i_r + h_r);
        float z = sigmoid_f(i_z + h_z);
        float n = tanhf(i_n + r * h_n);

        int sid = ids[t];
        if (last_pos[sid] == t) {     // last occurrence wins; rows are distinct
            float h_old = speaker_memory[(size_t)sid * (size_t)H + c];
            out[(size_t)sid * (size_t)H + c] = (1.0f - z) * n + z * h_old;
        }
    }
}

// ---------------------------------------------------------------------------
// Host-side launcher
// ---------------------------------------------------------------------------
static inline size_t align256(size_t x) { return (x + 255) & ~(size_t)255; }

extern "C" void kernel_launch(void* const* d_in, const int* in_sizes, int n_in,
                              void* d_out, int out_size, void* d_ws, size_t ws_size,
                              hipStream_t stream) {
    const float* speaker_memory = (const float*)d_in[0];
    const int*   speaker_ids    = (const int*)d_in[1];
    const float* edu_reps       = (const float*)d_in[2];
    const float* W_ih           = (const float*)d_in[3];
    const float* W_hh           = (const float*)d_in[4];
    const float* b_ih           = (const float*)d_in[5];
    const float* b_hh           = (const float*)d_in[6];
    float*       out            = (float*)d_out;

    const int H = in_sizes[5] / 3;          // 1024
    const int T = in_sizes[1];              // 8192
    const int S = in_sizes[0] / H;          // 65536

    // Workspace layout
    char* ws = (char*)d_ws;
    size_t off = 0;
    int*    last_pos = (int*)(ws + off);   off = align256(off + (size_t)S * sizeof(int));
    __bf16* Xbf      = (__bf16*)(ws + off); off = align256(off + (size_t)T * H * sizeof(__bf16));
    __bf16* Hbf      = (__bf16*)(ws + off); off = align256(off + (size_t)T * H * sizeof(__bf16));
    __bf16* Wihbf    = (__bf16*)(ws + off); off = align256(off + (size_t)3 * H * H * sizeof(__bf16));
    __bf16* Whhbf    = (__bf16*)(ws + off); off = align256(off + (size_t)3 * H * H * sizeof(__bf16));
    (void)ws_size; (void)n_in; (void)out_size;

    // 1) Output base = copy of speaker_memory (untouched rows pass through)
    hipMemcpyAsync(out, speaker_memory, (size_t)S * H * sizeof(float),
                   hipMemcpyDeviceToDevice, stream);

    // 2) last-occurrence positions
    init_lastpos_kernel<<<(S + 255) / 256, 256, 0, stream>>>(last_pos, S);
    scatter_lastpos_kernel<<<(T + 255) / 256, 256, 0, stream>>>(speaker_ids, last_pos, T);

    // 3) bf16 operand staging
    {
        int nX = (T * H) / 8;
        cvt_f32_to_bf16_kernel<<<(nX + 255) / 256, 256, 0, stream>>>(edu_reps, Xbf, nX);
        int nW = (3 * H * H) / 8;
        cvt_f32_to_bf16_kernel<<<(nW + 255) / 256, 256, 0, stream>>>(W_ih, Wihbf, nW);
        cvt_f32_to_bf16_kernel<<<(nW + 255) / 256, 256, 0, stream>>>(W_hh, Whhbf, nW);
        gather_h_bf16_kernel<<<T, 256, 0, stream>>>(speaker_memory, speaker_ids, Hbf, H);
    }

    // 4) Fused dual-GEMM (bf16 WMMA, f32 accum) + GRU + scatter
    dim3 grid(H / 64, T / 32);
    gru_fused_kernel<<<grid, 192, 0, stream>>>(Xbf, Hbf, Wihbf, Whhbf,
                                               b_ih, b_hh,
                                               speaker_memory, speaker_ids,
                                               last_pos, out, H);
}
